// TmpHierRMCRNNCell_9294309228710
// MI455X (gfx1250) — matrix operand order repaired and checked by reading, not compiled, not run on hardware
//
#include <hip/hip_runtime.h>
#include <hip/hip_bf16.h>
#include <math.h>

// ---------------------------------------------------------------------------
// Types for CDNA5 WMMA
// ---------------------------------------------------------------------------
typedef __bf16 bf16;
typedef __attribute__((ext_vector_type(16))) __bf16 v16bf;
typedef __attribute__((ext_vector_type(8)))  __bf16 v8bf;
typedef __attribute__((ext_vector_type(8)))  float  v8f;

// Model constants
#define BB   32
#define TT   128
#define UU   1024
#define HH   16
#define DHD  64
#define MM   8
#define LL   9      // M + 1
#define RR   17     // 2M + 1

// ---------------------------------------------------------------------------
// Helpers
// ---------------------------------------------------------------------------
__device__ __forceinline__ float gelu_tanh(float x) {
    float x3 = x * x * x;
    return 0.5f * x * (1.f + tanhf(0.7978845608028654f * (x + 0.044715f * x3)));
}

__device__ __forceinline__ float hard_sigmoid(float x) {
    return fminf(fmaxf(0.2f * x + 0.5f, 0.f), 1.f);
}

// A fragment load: two 16B runs per lane per ISA 16-bit A layout
__device__ __forceinline__ v16bf load_a_frag(const bf16* __restrict__ p) {
    v8bf a0 = *(const v8bf*)p;
    v8bf a1 = *(const v8bf*)(p + 16);
    v16bf a;
    #pragma unroll
    for (int i = 0; i < 8; i++) { a[i] = a0[i]; a[8 + i] = a1[i]; }
    return a;
}

// block-wide reduce of two partial sums; blockDim.x == 256
__device__ __forceinline__ void blockReduce2(float& s, float& q) {
    __shared__ float sh0[256];
    __shared__ float sh1[256];
    int t = threadIdx.x;
    sh0[t] = s; sh1[t] = q;
    __syncthreads();
    for (int off = 128; off > 0; off >>= 1) {
        if (t < off) { sh0[t] += sh0[t + off]; sh1[t] += sh1[t + off]; }
        __syncthreads();
    }
    s = sh0[0]; q = sh1[0];
    __syncthreads();
}

// ---------------------------------------------------------------------------
// One-time prep kernels
// ---------------------------------------------------------------------------

// dst[n*K + k] = (bf16) src[k*ldsrc + col0 + n]   (LDS-tiled transpose + cast)
__global__ void __launch_bounds__(256)
transpose_cast_kernel(const float* __restrict__ src, bf16* __restrict__ dst,
                      int K, int N, int ldsrc, int col0) {
    __shared__ float tile[32][33];
    int bk = blockIdx.x * 32;   // k-tile
    int bn = blockIdx.y * 32;   // n-tile
    int tx = threadIdx.x & 31;
    int ty = threadIdx.x >> 5;  // 0..7
    #pragma unroll
    for (int i = ty; i < 32; i += 8) {
        int k = bk + i, n = bn + tx;
        tile[i][tx] = (k < K && n < N) ? src[(size_t)k * ldsrc + col0 + n] : 0.f;
    }
    __syncthreads();
    #pragma unroll
    for (int i = ty; i < 32; i += 8) {
        int n = bn + i, k = bk + tx;
        if (n < N && k < K) dst[(size_t)n * K + k] = (bf16)tile[tx][i];
    }
}

// plain f32 -> bf16 cast
__global__ void cast_bf16_kernel(const float* __restrict__ src,
                                 bf16* __restrict__ dst, int n) {
    int i = blockIdx.x * blockDim.x + threadIdx.x;
    if (i < n) dst[i] = (bf16)src[i];
}

// bias1[3072] = [input_bias(1024), 0(2048)]
// bias2[5120] = [attention_bias(3072), recurrent_bias(2048)]
__global__ void bias_pack_kernel(const float* __restrict__ ib,
                                 const float* __restrict__ ab,
                                 const float* __restrict__ rb,
                                 float* __restrict__ b1, float* __restrict__ b2) {
    int j = blockIdx.x * blockDim.x + threadIdx.x;
    if (j < 3072) b1[j] = (j < 1024) ? ib[j] : 0.f;
    if (j < 5120) b2[j] = (j < 3072) ? ab[j] : rb[j - 3072];
}

// sinusoidal relative-position embedding table [17, 1024] (fp64, like ref)
__global__ void relemb_kernel(float* __restrict__ emb) {
    int idx = blockIdx.x * blockDim.x + threadIdx.x;
    if (idx >= RR * UU) return;
    int i  = idx & (UU - 1);
    int rj = idx >> 10;
    double p    = (double)(rj - MM);
    double expo = (double)(i & ~1) / (double)UU;
    double ang  = p / pow(10000.0, expo);
    emb[idx] = (float)((i & 1) ? cos(ang) : sin(ang));
}

// rr[17,1024] = emb[17,1024] @ rel_kernel[1024,1024]
__global__ void relproj_kernel(const float* __restrict__ emb,
                               const float* __restrict__ rk,
                               float* __restrict__ rr) {
    int idx = blockIdx.x * blockDim.x + threadIdx.x;
    if (idx >= RR * UU) return;
    int u  = idx & (UU - 1);
    int rj = idx >> 10;
    const float* e = emb + (size_t)rj * UU;
    float acc = 0.f;
    for (int k = 0; k < UU; k++) acc += e[k] * rk[(size_t)k * UU + u];
    rr[idx] = acc;
}

__global__ void zero_kernel(float* __restrict__ p, int n) {
    int i = blockIdx.x * blockDim.x + threadIdx.x;
    if (i < n) p[i] = 0.f;
}

// ---------------------------------------------------------------------------
// WMMA GEMM: C[M,N] = A[M,K](bf16,row,lda) @ Bt[N,K](bf16,row)^T + bias[N]
// One wave computes a 16 x (16*NBLK) strip; A fragment reused NBLK times.
// Double-buffered K pipeline: next iteration's fragments are loaded before
// the current WMMAs issue, so waits are partial instead of full drains.
// EPI=0: store f32.  EPI=1: gelu then store bf16.
// ---------------------------------------------------------------------------
template <int EPI, int NBLK>
__global__ void __launch_bounds__(256)
gemm_wmma_kernel(const bf16* __restrict__ A, int lda,
                 const bf16* __restrict__ Bt,
                 const float* __restrict__ bias,
                 void* __restrict__ Cout,
                 int Mrows, int N, int K) {
    const int lane = threadIdx.x & 31;
    const int wid  = (blockIdx.x * blockDim.x + threadIdx.x) >> 5;
    const int NTG  = N / (16 * NBLK);
    const int MT   = Mrows >> 4;
    if (wid >= MT * NTG) return;          // wave-uniform: EXEC stays all-ones
    const int tm  = (wid / NTG) << 4;
    const int tn  = (wid % NTG) * (16 * NBLK);
    const int r16 = lane & 15;            // A-row within tile / B-column within tile
    const int hi  = lane >> 4;            // half-select per ISA fragment layout

    // A fragment: lanes 0-15 rows M=0..15 hold K=k0..k0+15, lanes 16-31 K=k0+16..k0+31
    const bf16* Arow = A + (size_t)(tm + r16) * lda + hi * 16;
    // B fragment: column n = lane&15; lanes 0-15 hold K=k0..k0+15, lanes 16-31 K=k0+16..k0+31
    const bf16* Brow[NBLK];
    #pragma unroll
    for (int j = 0; j < NBLK; j++)
        Brow[j] = Bt + (size_t)(tn + 16 * j + r16) * K + hi * 16;

    v8f acc[NBLK];
    #pragma unroll
    for (int j = 0; j < NBLK; j++) { v8f z = {}; acc[j] = z; }

    // ---- software-pipelined K loop (double buffered) ----
    v16bf a_cur = *(const v16bf*)(Arow);
    v16bf b_cur[NBLK];
    #pragma unroll
    for (int j = 0; j < NBLK; j++) b_cur[j] = *(const v16bf*)(Brow[j]);

    #pragma unroll 2
    for (int k0 = 0; k0 < K - 32; k0 += 32) {
        // issue next iteration's loads first
        v16bf a_nxt = *(const v16bf*)(Arow + k0 + 32);
        v16bf b_nxt[NBLK];
        #pragma unroll
        for (int j = 0; j < NBLK; j++) b_nxt[j] = *(const v16bf*)(Brow[j] + k0 + 32);
        // compute on current fragments (waits only on older loads)
        #pragma unroll
        for (int j = 0; j < NBLK; j++)
            acc[j] = __builtin_amdgcn_wmma_f32_16x16x32_bf16(
                false, a_cur, false, b_cur[j], (short)0, acc[j], false, false);
        a_cur = a_nxt;
        #pragma unroll
        for (int j = 0; j < NBLK; j++) b_cur[j] = b_nxt[j];
    }
    #pragma unroll
    for (int j = 0; j < NBLK; j++)
        acc[j] = __builtin_amdgcn_wmma_f32_16x16x32_bf16(
            false, a_cur, false, b_cur[j], (short)0, acc[j], false, false);

    // ---- epilogue ----
    #pragma unroll
    for (int j = 0; j < NBLK; j++) {
        const int col = tn + 16 * j + r16;
        const float bv = bias[col];
        #pragma unroll
        for (int e = 0; e < 8; e++) {
            int row = tm + e + (hi << 3);  // C/D layout: vgpr e -> M=e (lanes<16) / M=e+8
            float v = acc[j][e] + bv;
            if (EPI == 1) ((bf16*)Cout)[(size_t)row * N + col] = (bf16)gelu_tanh(v);
            else          ((float*)Cout)[(size_t)row * N + col] = v;
        }
    }
}

// ---------------------------------------------------------------------------
// Per-step elementwise / attention kernels
// ---------------------------------------------------------------------------

// mp slot 8 (current input row) <- xp = s1out[:, :1024]  (f32 + bf16 shadow)
__global__ void copyxp_kernel(const float* __restrict__ s1,
                              float* __restrict__ mp, bf16* __restrict__ mpb) {
    int idx = blockIdx.x * blockDim.x + threadIdx.x;
    if (idx >= BB * UU) return;
    int b = idx >> 10, j = idx & (UU - 1);
    float v = s1[(size_t)b * 3072 + j];
    size_t o = (size_t)(b * LL + MM) * UU + j;
    mp[o]  = v;
    mpb[o] = (bf16)v;
}

// one thread per (b, h, qi): 9-key softmax attention with relative positions.
// scores[qi][ki] = SCALE * q . (k[ki] + r[qi-ki+8])
__global__ void __launch_bounds__(256)
attn_kernel(const float* __restrict__ s2, const float* __restrict__ rr,
            float* __restrict__ att) {
    int idx = blockIdx.x * blockDim.x + threadIdx.x;
    if (idx >= BB * HH * LL) return;
    int qi = idx % LL;
    int hh = (idx / LL) & (HH - 1);
    int b  = idx / (LL * HH);
    const float* base = s2 + (size_t)b * LL * 5120;
    const float* q    = base + (size_t)qi * 5120 + hh * DHD;

    float qv[DHD];
    #pragma unroll
    for (int d = 0; d < DHD; d++) qv[d] = q[d];

    float sc[LL];
    float mx = -1e30f;
    #pragma unroll
    for (int ki = 0; ki < LL; ki++) {
        const float* kp = base + (size_t)ki * 5120 + 1024 + hh * DHD;
        const float* rp = rr + (size_t)(qi - ki + MM) * UU + hh * DHD;
        float acc = 0.f;
        #pragma unroll
        for (int d = 0; d < DHD; d++) acc += qv[d] * (kp[d] + rp[d]);
        sc[ki] = acc * 0.125f;            // 1/sqrt(64)
        mx = fmaxf(mx, sc[ki]);
    }
    float ssum = 0.f;
    #pragma unroll
    for (int ki = 0; ki < LL; ki++) { sc[ki] = expf(sc[ki] - mx); ssum += sc[ki]; }
    float inv = 1.f / ssum;

    float* op = att + (size_t)(b * LL + qi) * UU + hh * DHD;
    #pragma unroll 4
    for (int d = 0; d < DHD; d++) {
        float a = 0.f;
        #pragma unroll
        for (int ki = 0; ki < LL; ki++)
            a += sc[ki] * base[(size_t)ki * 5120 + 2048 + hh * DHD + d];
        op[d] = a * inv;
    }
}

// a1 = LayerNorm(mp + att) * gamma + beta   (f32 out + bf16 shadow)
__global__ void __launch_bounds__(256)
ln_add_kernel(const float* __restrict__ xa, const float* __restrict__ xb,
              const float* __restrict__ g, const float* __restrict__ be,
              float* __restrict__ out, bf16* __restrict__ outb) {
    int row = blockIdx.x;
    const float* pa = xa + (size_t)row * UU;
    const float* pb = xb + (size_t)row * UU;
    float v[4], s = 0.f, q = 0.f;
    #pragma unroll
    for (int i = 0; i < 4; i++) {
        int j = threadIdx.x + i * 256;
        v[i] = pa[j] + pb[j];
        s += v[i]; q += v[i] * v[i];
    }
    blockReduce2(s, q);
    float mean = s * (1.f / UU);
    float var  = q * (1.f / UU) - mean * mean;
    float rs   = rsqrtf(var + 1e-6f);
    #pragma unroll
    for (int i = 0; i < 4; i++) {
        int j = threadIdx.x + i * 256;
        float o = (v[i] - mean) * rs * g[j] + be[j];
        out[(size_t)row * UU + j]  = o;
        outb[(size_t)row * UU + j] = (bf16)o;
    }
}

// cand = LN(a1+m2); gates from s1out (xt@Wg) + s2out (mp@Wr + bias);
// nmp = hs(gf+1)*mp + hs(gi)*tanh(cand); slots 0..7 -> mp/mpb, slot 8 -> y[t]
__global__ void __launch_bounds__(256)
finalize_kernel(const float* __restrict__ a1, const float* __restrict__ m2,
                const float* __restrict__ s1, const float* __restrict__ s2,
                const float* __restrict__ g2, const float* __restrict__ b2,
                float* __restrict__ mp, bf16* __restrict__ mpb,
                float* __restrict__ out, int t) {
    int row = blockIdx.x;                  // 0..287
    int b = row / LL, l = row - b * LL;
    const float* pa = a1 + (size_t)row * UU;
    const float* pm = m2 + (size_t)row * UU;
    float v[4], s = 0.f, q = 0.f;
    #pragma unroll
    for (int i = 0; i < 4; i++) {
        int j = threadIdx.x + i * 256;
        v[i] = pa[j] + pm[j];
        s += v[i]; q += v[i] * v[i];
    }
    blockReduce2(s, q);
    float mean = s * (1.f / UU);
    float var  = q * (1.f / UU) - mean * mean;
    float rs   = rsqrtf(var + 1e-6f);
    #pragma unroll
    for (int i = 0; i < 4; i++) {
        int j = threadIdx.x + i * 256;
        float cand = (v[i] - mean) * rs * g2[j] + b2[j];
        float gi = s1[(size_t)b * 3072 + 1024 + j] + s2[(size_t)row * 5120 + 3072 + j];
        float gf = s1[(size_t)b * 3072 + 2048 + j] + s2[(size_t)row * 5120 + 4096 + j] + 1.0f;
        gi = hard_sigmoid(gi);
        gf = hard_sigmoid(gf);
        float nv = gf * mp[(size_t)row * UU + j] + gi * tanhf(cand);
        if (l < MM) {
            mp[(size_t)row * UU + j]  = nv;
            mpb[(size_t)row * UU + j] = (bf16)nv;
        } else {
            out[((size_t)b * TT + t) * UU + j] = nv;
        }
    }
}

// ---------------------------------------------------------------------------
// Host orchestration
// ---------------------------------------------------------------------------
extern "C" void kernel_launch(void* const* d_in, const int* in_sizes, int n_in,
                              void* d_out, int out_size, void* d_ws, size_t ws_size,
                              hipStream_t stream) {
    (void)in_sizes; (void)n_in; (void)out_size; (void)ws_size;

    const float* x     = (const float*)d_in[0];   // [32,128,1024]
    const float* Wi    = (const float*)d_in[1];   // [1024,1024]
    const float* bi    = (const float*)d_in[2];   // [1024]
    const float* Wg    = (const float*)d_in[3];   // [1024,2048]
    const float* Wr    = (const float*)d_in[4];   // [1024,2048]
    const float* br    = (const float*)d_in[5];   // [2048]
    const float* Wa    = (const float*)d_in[6];   // [1024,3072]
    const float* ba    = (const float*)d_in[7];   // [3072]
    const float* Wm    = (const float*)d_in[8];   // [1024,2048]
    const float* bm    = (const float*)d_in[9];   // [2048]
    const float* lng   = (const float*)d_in[10];  // [2048]
    const float* lnb   = (const float*)d_in[11];  // [2048]
    const float* Wrel  = (const float*)d_in[12];  // [1024,1024]
    float* out = (float*)d_out;                   // [32,128,1024]

    // workspace carve-up
    char* base = (char*)d_ws;
    size_t off = 0;
    auto take = [&](size_t bytes) -> void* {
        void* r = base + off;
        off = (off + bytes + 255) & ~(size_t)255;
        return r;
    };
    bf16*  WiGt  = (bf16*)take((size_t)3072 * 1024 * 2);   // [Wi|Wg]^T bf16
    bf16*  WaRt  = (bf16*)take((size_t)5120 * 1024 * 2);   // [Wa|Wr]^T bf16
    bf16*  W1t   = (bf16*)take((size_t)1024 * 1024 * 2);   // mlp[:, :U]^T
    bf16*  W2t   = (bf16*)take((size_t)1024 * 1024 * 2);   // mlp[:, U:]^T
    bf16*  xall  = (bf16*)take((size_t)BB * TT * UU * 2);  // x cast to bf16 once
    float* bias1 = (float*)take(3072 * 4);
    float* bias2 = (float*)take(5120 * 4);
    float* emb   = (float*)take((size_t)RR * UU * 4);
    float* rrel  = (float*)take((size_t)RR * UU * 4);
    float* mp    = (float*)take((size_t)BB * LL * UU * 4); // persistent memory + xp slot
    bf16*  mpb   = (bf16*)take((size_t)BB * LL * UU * 2);  // bf16 shadow of mp
    float* s1o   = (float*)take((size_t)BB * 3072 * 4);
    float* s2o   = (float*)take((size_t)BB * LL * 5120 * 4);
    float* att   = (float*)take((size_t)BB * LL * UU * 4);
    float* a1    = (float*)take((size_t)BB * LL * UU * 4);
    bf16*  a1b   = (bf16*)take((size_t)BB * LL * UU * 2);  // bf16 shadow of a1
    bf16*  hbuf  = (bf16*)take((size_t)BB * LL * UU * 2);  // gelu output, bf16
    float* m2    = (float*)take((size_t)BB * LL * UU * 4);

    // ---- one-time prep ----
    transpose_cast_kernel<<<dim3(32, 32), 256, 0, stream>>>(Wi, WiGt, 1024, 1024, 1024, 0);
    transpose_cast_kernel<<<dim3(32, 64), 256, 0, stream>>>(Wg, WiGt + (size_t)1024 * 1024, 1024, 2048, 2048, 0);
    transpose_cast_kernel<<<dim3(32, 96), 256, 0, stream>>>(Wa, WaRt, 1024, 3072, 3072, 0);
    transpose_cast_kernel<<<dim3(32, 64), 256, 0, stream>>>(Wr, WaRt + (size_t)3072 * 1024, 1024, 2048, 2048, 0);
    transpose_cast_kernel<<<dim3(32, 32), 256, 0, stream>>>(Wm, W1t, 1024, 1024, 2048, 0);
    transpose_cast_kernel<<<dim3(32, 32), 256, 0, stream>>>(Wm, W2t, 1024, 1024, 2048, 1024);
    cast_bf16_kernel<<<(BB * TT * UU + 255) / 256, 256, 0, stream>>>(x, xall, BB * TT * UU);
    bias_pack_kernel<<<(5120 + 255) / 256, 256, 0, stream>>>(bi, ba, br, bias1, bias2);
    relemb_kernel<<<(RR * UU + 255) / 256, 256, 0, stream>>>(emb);
    relproj_kernel<<<(RR * UU + 255) / 256, 256, 0, stream>>>(emb, Wrel, rrel);
    zero_kernel<<<(BB * LL * UU + 255) / 256, 256, 0, stream>>>(mp, BB * LL * UU);
    zero_kernel<<<(BB * LL * UU / 2 + 255) / 256, 256, 0, stream>>>((float*)mpb, BB * LL * UU / 2);

    // ---- sequential scan over T steps ----
    for (int t = 0; t < TT; t++) {
        const bf16* xt = xall + (size_t)t * UU;   // row b at stride T*U

        // S1: [32,3072] = xt @ [Wi|Wg]  (+[input_bias|0])     2*48=96 waves
        gemm_wmma_kernel<0, 4><<<12, 256, 0, stream>>>(xt, TT * UU, WiGt, bias1, s1o, 32, 3072, 1024);
        copyxp_kernel<<<(BB * UU + 255) / 256, 256, 0, stream>>>(s1o, mp, mpb);

        // S2: [288,5120] = mp @ [Wa|Wr]  (+[attn_bias|rec_bias])  18*80=1440 waves
        gemm_wmma_kernel<0, 4><<<180, 256, 0, stream>>>(mpb, UU, WaRt, bias2, s2o, 288, 5120, 1024);

        // attention + residual LN
        attn_kernel<<<(BB * HH * LL + 255) / 256, 256, 0, stream>>>(s2o, rrel, att);
        ln_add_kernel<<<288, 256, 0, stream>>>(mp, att, lng, lnb, a1, a1b);

        // MLP: h = gelu(a1 @ W1 + b1) ; m2 = h @ W2 + b2      18*32=576 waves each
        gemm_wmma_kernel<1, 2><<<72, 256, 0, stream>>>(a1b, UU, W1t, bm, hbuf, 288, 1024, 1024);
        gemm_wmma_kernel<0, 2><<<72, 256, 0, stream>>>(hbuf, UU, W2t, bm + 1024, m2, 288, 1024, 1024);

        // cand LN + LSTM-style gating; updates mp slots 0..7 and emits y[:,t,:]
        finalize_kernel<<<288, 256, 0, stream>>>(a1, m2, s1o, s2o,
                                                 lng + 1024, lnb + 1024, mp, mpb, out, t);
    }
}
